// GRU_D_75531294868001
// MI455X (gfx1250) — matrix operand order repaired and checked by reading, not compiled, not run on hardware
//
#include <hip/hip_runtime.h>
#include <hip/hip_bf16.h>

typedef __bf16 bf16_t;
typedef __attribute__((ext_vector_type(16))) __bf16 v16bf;
typedef __attribute__((ext_vector_type(8)))  __bf16 v8bf;
typedef __attribute__((ext_vector_type(8)))  float   v8f;

#define TT 256
#define EE 1024
#define BB 2048
#define DD 32
#define SS 2
#define HH 1024
#define DINN 64   // D*S

// ---------------------------------------------------------------------------
// Elementwise / setup kernels
// ---------------------------------------------------------------------------
__global__ void k_zero_f32(float* p, int n) {
  int i = blockIdx.x * blockDim.x + threadIdx.x;
  if (i < n) p[i] = 0.0f;
}

// src: K x N row-major f32  ->  dst: N x K row-major bf16 (transposed weight)
__global__ void k_transpose_to_bf16(const float* __restrict__ src, bf16_t* __restrict__ dst,
                                    int K, int N) {
  int i = blockIdx.x * blockDim.x + threadIdx.x;
  if (i < N * K) {
    int n = i / K, k = i % K;
    dst[i] = (bf16_t)src[(size_t)k * N + n];
  }
}

// W2: H x 31 -> W2t: 32 x H (bf16, row 31 zero)
__global__ void k_build_w2t(const float* __restrict__ W2, bf16_t* __restrict__ dst) {
  int i = blockIdx.x * blockDim.x + threadIdx.x;
  if (i < 32 * HH) {
    int n = i / HH, k = i % HH;
    dst[i] = (n < DD - 1) ? (bf16_t)W2[(size_t)k * (DD - 1) + n] : (bf16_t)0.0f;
  }
}

__global__ void k_build_b2pad(const float* __restrict__ b2, float* __restrict__ dst) {
  int i = threadIdx.x;
  if (i < 32) dst[i] = (i < DD - 1) ? b2[i] : 0.0f;
}

// [Wx;Vm] (128 x 3H) transposed -> 3H x 128 bf16
__global__ void k_build_wxvm(const float* __restrict__ Wx, const float* __restrict__ Vm,
                             bf16_t* __restrict__ dst) {
  int i = blockIdx.x * blockDim.x + threadIdx.x;
  if (i < 3 * HH * 128) {
    int n = i / 128, k = i % 128;
    float v = (k < DINN) ? Wx[(size_t)k * (3 * HH) + n]
                         : Vm[(size_t)(k - DINN) * (3 * HH) + n];
    dst[i] = (bf16_t)v;
  }
}

__global__ void k_gather_h(const float* __restrict__ h, const int* __restrict__ idx,
                           bf16_t* __restrict__ Ah) {
  int i = blockIdx.x * blockDim.x + threadIdx.x;
  if (i < EE * HH) {
    int e = i / HH, n = i % HH;
    Ah[i] = (bf16_t)h[(size_t)idx[e] * HH + n];
  }
}

// ---------------------------------------------------------------------------
// Loss accumulation (block LDS reduction -> 4 global atomics per block)
// ---------------------------------------------------------------------------
__global__ void k_loss(const float* __restrict__ P, const float* __restrict__ X,
                       const float* __restrict__ M, const float* __restrict__ obs,
                       int step, float* __restrict__ acc) {
  __shared__ float s0[256], s1[256], s2[256], s3[256];
  int tid = threadIdx.x;
  int i = blockIdx.x * 256 + tid;
  float tcur = obs[step];
  float valid = (tcur > 0.0f) ? 1.0f : 0.0f;
  float a0 = 0.f, a1 = 0.f, a2 = 0.f, a3 = 0.f;
  if (i < EE * (DD - 1)) {
    int e = i / (DD - 1), j = i % (DD - 1);
    size_t eb = (size_t)step * EE + e;
    float Xo = X[(eb * DD + (j + 1)) * SS];       // X[t,e,j+1,0]
    float Mo = M[eb * DD + (j + 1)];
    float diff = Xo - P[e * 32 + j];
    a0 = valid * diff * diff * Mo;
    a1 = valid * fabsf(diff) * Mo;
    a2 = valid * fabsf(diff) / (Xo + 1e-8f) * Mo;
    a3 = valid * Mo;
  }
  s0[tid] = a0; s1[tid] = a1; s2[tid] = a2; s3[tid] = a3;
  __syncthreads();
  for (int st = 128; st > 0; st >>= 1) {
    if (tid < st) {
      s0[tid] += s0[tid + st]; s1[tid] += s1[tid + st];
      s2[tid] += s2[tid + st]; s3[tid] += s3[tid + st];
    }
    __syncthreads();
  }
  if (tid == 0) {
    atomicAdd(&acc[0], s0[0]); atomicAdd(&acc[1], s1[0]);
    atomicAdd(&acc[2], s2[0]); atomicAdd(&acc[3], s3[0]);
  }
}

// ---------------------------------------------------------------------------
// Observation step: update last_x/last_t, emit interval (bf16), column sums
// ---------------------------------------------------------------------------
__global__ void k_observe(const float* __restrict__ X, const float* __restrict__ M,
                          const int* __restrict__ idx, const float* __restrict__ obs, int step,
                          float* __restrict__ last_x, float* __restrict__ last_t,
                          bf16_t* __restrict__ IntB, float* __restrict__ colsum) {
  __shared__ float sX[DINN], sM[DINN];
  int tid = threadIdx.x;
  int k = tid & (DINN - 1);
  int e = blockIdx.x * 4 + (tid >> 6);
  if (tid < DINN) { sX[tid] = 0.0f; sM[tid] = 0.0f; }
  __syncthreads();
  size_t eb = (size_t)step * EE + e;
  float m2 = M[eb * DD + (k >> 1)];
  float xf = X[eb * DINN + k];
  int row = idx[e];
  size_t ro = (size_t)row * DINN + k;
  float lx = last_x[ro] * (1.0f - m2) + xf * m2;
  last_x[ro] = lx;
  float tcur = obs[step];
  float lt = last_t[ro];
  float interval = tcur - lt;                 // uses OLD last_t (matches reference)
  last_t[ro] = lt * (1.0f - m2) + tcur * m2;
  IntB[(size_t)e * DINN + k] = (bf16_t)interval;
  atomicAdd(&sX[k], xf);
  atomicAdd(&sM[k], m2);
  __syncthreads();
  if (tid < DINN) {
    atomicAdd(&colsum[tid], sX[tid]);
    atomicAdd(&colsum[DINN + tid], sM[tid]);
  }
}

// X_hat assembly: AXM = [X_hat | M2] as bf16 (E x 128) for the fused Wx/Vm GEMM
__global__ void k_xhat(const float* __restrict__ X, const float* __restrict__ M,
                       const int* __restrict__ idx, int step,
                       const float* __restrict__ last_x, const float* __restrict__ Gx,
                       const float* __restrict__ colsum, bf16_t* __restrict__ AXM) {
  int i = blockIdx.x * blockDim.x + threadIdx.x;
  if (i < EE * DINN) {
    int e = i / DINN, k = i % DINN;
    size_t eb = (size_t)step * EE + e;
    float m2 = M[eb * DD + (k >> 1)];
    float xf = X[eb * DINN + k];
    float gx = Gx[i];
    float lx = last_x[(size_t)idx[e] * DINN + k];   // already updated == lx
    float mean = colsum[k] / (colsum[DINN + k] + (float)EE * 1e-6f);
    float xh = m2 * xf + (1.0f - m2) * gx * lx + (1.0f - m2) * (1.0f - gx) * mean;
    AXM[(size_t)e * 128 + k] = (bf16_t)xh;
    AXM[(size_t)e * 128 + DINN + k] = (bf16_t)m2;
  }
}

// hs = gamma_h * h[idx]  (f32 master + bf16 for WMMA A)
__global__ void k_scale_h(const float* __restrict__ h, const int* __restrict__ idx,
                          const float* __restrict__ Gh, float* __restrict__ hsF,
                          bf16_t* __restrict__ hsB) {
  int i = blockIdx.x * blockDim.x + threadIdx.x;
  if (i < EE * HH) {
    int e = i / HH, n = i % HH;
    float v = h[(size_t)idx[e] * HH + n] * Gh[i];
    hsF[i] = v; hsB[i] = (bf16_t)v;
  }
}

__global__ void k_rh(const float* __restrict__ RZ, const float* __restrict__ hsF,
                     bf16_t* __restrict__ rh) {
  int i = blockIdx.x * blockDim.x + threadIdx.x;
  if (i < EE * HH) {
    int e = i / HH, n = i % HH;
    rh[i] = (bf16_t)(RZ[(size_t)e * 2 * HH + n] * hsF[i]);   // r * h'
  }
}

__global__ void k_hupdate(const float* __restrict__ RZ, const float* __restrict__ hsF,
                          const float* __restrict__ HT, const int* __restrict__ idx,
                          float* __restrict__ h) {
  int i = blockIdx.x * blockDim.x + threadIdx.x;
  if (i < EE * HH) {
    int e = i / HH, n = i % HH;
    float z = RZ[(size_t)e * 2 * HH + HH + n];
    h[(size_t)idx[e] * HH + n] = (1.0f - z) * hsF[i] + z * HT[i];
  }
}

__global__ void k_finalize(const float* __restrict__ acc, float* __restrict__ out) {
  if (threadIdx.x == 0) {
    float tot = acc[3];
    out[0] = acc[0] / tot;
    out[1] = acc[1] / tot;
    out[2] = acc[2] / tot;
  }
}

// ---------------------------------------------------------------------------
// WMMA GEMM: C(MxN,f32 acc) = A(MxK,bf16 row-major) * Bt(NxK,bf16 row-major)^T
// One wave per 32 x (16*NT) macro-tile. K in steps of 32, double-buffered and
// unrolled by 2 (K % 64 == 0) so buffer indices are compile-time and next
// iteration's loads overlap the current WMMAs.
// ACT: 0=none 1=relu 2=sigmoid 3=tanh 4=exp(-relu)
// ---------------------------------------------------------------------------
template <int NT>
__device__ __forceinline__ void wg_load(const bf16_t* const* aptr, const bf16_t* const* bptr,
                                        int koff, v8bf* alo, v8bf* ahi, v16bf* bv) {
#pragma unroll
  for (int mi = 0; mi < 2; ++mi) {
    alo[mi] = *(const v8bf*)(aptr[mi] + koff);
    ahi[mi] = *(const v8bf*)(aptr[mi] + koff + 8);
  }
#pragma unroll
  for (int ni = 0; ni < NT; ++ni)
    bv[ni] = *(const v16bf*)(bptr[ni] + koff);
}

template <int NT>
__device__ __forceinline__ void wg_compute(const v8bf* alo, const v8bf* ahi,
                                           const v16bf* bv, v8f acc[2][NT]) {
  v16bf a[2];
#pragma unroll
  for (int mi = 0; mi < 2; ++mi)
#pragma unroll
    for (int j = 0; j < 8; ++j) { a[mi][j] = alo[mi][j]; a[mi][j + 8] = ahi[mi][j]; }
#pragma unroll
  for (int ni = 0; ni < NT; ++ni)
#pragma unroll
    for (int mi = 0; mi < 2; ++mi)
      acc[mi][ni] = __builtin_amdgcn_wmma_f32_16x16x32_bf16(
          false, a[mi], false, bv[ni], (short)0, acc[mi][ni], false, false);
}

template <int ACT, int OUTB, int NT>
__global__ void __launch_bounds__(32)
k_wmma_gemm(const bf16_t* __restrict__ A, const bf16_t* __restrict__ Bt,
            const float* __restrict__ bias, const float* __restrict__ addsrc, int ldadd,
            float* __restrict__ outF, bf16_t* __restrict__ outB, int N, int K) {
  const int lane = threadIdx.x;                 // wave32
  const int m0 = blockIdx.y * 32;
  const int n0 = blockIdx.x * (16 * NT);
  const int am  = lane & 15;
  const int akb = (lane >> 4) << 4;             // A: lanes 16-31 hold K 16..31
  const int bn  = lane & 15;
  const int bkb = (lane >> 4) << 4;             // B: lanes 16-31 hold K 16..31

  const bf16_t* aptr[2];
  aptr[0] = A + (size_t)(m0 + am) * K + akb;
  aptr[1] = aptr[0] + (size_t)16 * K;
  const bf16_t* bptr[NT];
#pragma unroll
  for (int ni = 0; ni < NT; ++ni)
    bptr[ni] = Bt + (size_t)(n0 + ni * 16 + bn) * K + bkb;

  v8f acc[2][NT];
#pragma unroll
  for (int mi = 0; mi < 2; ++mi)
#pragma unroll
    for (int ni = 0; ni < NT; ++ni)
#pragma unroll
      for (int j = 0; j < 8; ++j) acc[mi][ni][j] = 0.0f;

  v8bf alo0[2], ahi0[2], alo1[2], ahi1[2];
  v16bf bv0[NT], bv1[NT];

  wg_load<NT>(aptr, bptr, 0, alo0, ahi0, bv0);          // prologue: k = 0
  for (int k0 = 0; k0 < K; k0 += 64) {                  // K % 64 == 0
    wg_load<NT>(aptr, bptr, k0 + 32, alo1, ahi1, bv1);  // fetch k0+32 (always valid)
    wg_compute<NT>(alo0, ahi0, bv0, acc);               // consume k0
    if (k0 + 64 < K)
      wg_load<NT>(aptr, bptr, k0 + 64, alo0, ahi0, bv0);
    wg_compute<NT>(alo1, ahi1, bv1, acc);               // consume k0+32
  }

  const int ncl = lane & 15;
  const int mofs = (lane >> 4) << 3;            // C: lanes 16-31 hold M rows 8..15
#pragma unroll
  for (int ni = 0; ni < NT; ++ni) {
    int ncol = n0 + ni * 16 + ncl;
    float bvl = bias ? bias[ncol] : 0.0f;
#pragma unroll
    for (int mi = 0; mi < 2; ++mi) {
      int mbase = m0 + mi * 16 + mofs;
#pragma unroll
      for (int r = 0; r < 8; ++r) {
        int mrow = mbase + r;
        float v = acc[mi][ni][r] + bvl;
        if (addsrc) v += addsrc[(size_t)mrow * ldadd + ncol];
        if (ACT == 1)      v = fmaxf(v, 0.0f);
        else if (ACT == 2) v = 1.0f / (1.0f + __expf(-v));
        else if (ACT == 3) v = tanhf(v);
        else if (ACT == 4) v = __expf(-fmaxf(v, 0.0f));
        if (OUTB) outB[(size_t)mrow * N + ncol] = (bf16_t)v;
        else      outF[(size_t)mrow * N + ncol] = v;
      }
    }
  }
}

// ---------------------------------------------------------------------------
extern "C" void kernel_launch(void* const* d_in, const int* in_sizes, int n_in,
                              void* d_out, int out_size, void* d_ws, size_t ws_size,
                              hipStream_t stream) {
  (void)in_sizes; (void)n_in; (void)out_size; (void)ws_size;
  const float* obs  = (const float*)d_in[0];
  const float* X    = (const float*)d_in[1];
  const float* M    = (const float*)d_in[2];
  const int*   bidx = (const int*)d_in[3];
  const float* Wx   = (const float*)d_in[4];
  const float* Vm   = (const float*)d_in[5];
  const float* U_rz = (const float*)d_in[6];
  const float* U_h  = (const float*)d_in[7];
  const float* b    = (const float*)d_in[8];
  const float* Wgx  = (const float*)d_in[9];
  const float* Wgh  = (const float*)d_in[10];
  const float* W1   = (const float*)d_in[11];
  const float* b1   = (const float*)d_in[12];
  const float* W2   = (const float*)d_in[13];
  const float* b2   = (const float*)d_in[14];
  float* out = (float*)d_out;

  char* base = (char*)d_ws;
  size_t off = 0;
  auto alloc = [&](size_t bytes) -> void* {
    void* p = base + off;
    off = (off + bytes + 255) & ~(size_t)255;
    return p;
  };
  float*  h      = (float*)alloc(4ull * BB * HH);
  float*  last_t = (float*)alloc(4ull * BB * DINN);
  float*  last_x = (float*)alloc(4ull * BB * DINN);
  float*  acc    = (float*)alloc(4ull * 4);
  float*  colsum = (float*)alloc(4ull * 2 * DINN);
  bf16_t* Ah     = (bf16_t*)alloc(2ull * EE * HH);
  bf16_t* Hrelu  = (bf16_t*)alloc(2ull * EE * HH);
  float*  P      = (float*)alloc(4ull * EE * 32);
  bf16_t* IntB   = (bf16_t*)alloc(2ull * EE * DINN);
  float*  Gx     = (float*)alloc(4ull * EE * DINN);
  float*  Gh     = (float*)alloc(4ull * EE * HH);
  bf16_t* AXM    = (bf16_t*)alloc(2ull * EE * 128);
  float*  PRE    = (float*)alloc(4ull * EE * 3 * HH);
  float*  hsF    = (float*)alloc(4ull * EE * HH);
  bf16_t* hsB    = (bf16_t*)alloc(2ull * EE * HH);
  float*  RZ     = (float*)alloc(4ull * EE * 2 * HH);
  bf16_t* rh     = (bf16_t*)alloc(2ull * EE * HH);
  float*  HT     = (float*)alloc(4ull * EE * HH);
  bf16_t* W1t    = (bf16_t*)alloc(2ull * HH * HH);
  bf16_t* W2t    = (bf16_t*)alloc(2ull * 32 * HH);
  bf16_t* Wgxt   = (bf16_t*)alloc(2ull * DINN * DINN);
  bf16_t* Wght   = (bf16_t*)alloc(2ull * HH * DINN);
  bf16_t* U_rzt  = (bf16_t*)alloc(2ull * 2 * HH * HH);
  bf16_t* U_ht   = (bf16_t*)alloc(2ull * HH * HH);
  bf16_t* WxVmt  = (bf16_t*)alloc(2ull * 3 * HH * 128);
  float*  b2pad  = (float*)alloc(4ull * 32);

  const int TPB = 256;
  auto gsz = [](long n, int tpb) { return (int)((n + tpb - 1) / tpb); };

  // ---- init carry + accumulators (harness poisons d_ws; must zero) ----
  k_zero_f32<<<gsz((long)BB * HH, TPB), TPB, 0, stream>>>(h, BB * HH);
  k_zero_f32<<<gsz((long)BB * DINN, TPB), TPB, 0, stream>>>(last_t, BB * DINN);
  k_zero_f32<<<gsz((long)BB * DINN, TPB), TPB, 0, stream>>>(last_x, BB * DINN);
  k_zero_f32<<<1, TPB, 0, stream>>>(acc, 4);

  // ---- one-time weight conversion (bf16, transposed to N x K) ----
  k_transpose_to_bf16<<<gsz((long)HH * HH, TPB), TPB, 0, stream>>>(W1, W1t, HH, HH);
  k_build_w2t<<<gsz(32l * HH, TPB), TPB, 0, stream>>>(W2, W2t);
  k_build_b2pad<<<1, 32, 0, stream>>>(b2, b2pad);
  k_transpose_to_bf16<<<gsz((long)DINN * DINN, TPB), TPB, 0, stream>>>(Wgx, Wgxt, DINN, DINN);
  k_transpose_to_bf16<<<gsz((long)HH * DINN, TPB), TPB, 0, stream>>>(Wgh, Wght, DINN, HH);
  k_transpose_to_bf16<<<gsz(2l * HH * HH, TPB), TPB, 0, stream>>>(U_rz, U_rzt, HH, 2 * HH);
  k_transpose_to_bf16<<<gsz((long)HH * HH, TPB), TPB, 0, stream>>>(U_h, U_ht, HH, HH);
  k_build_wxvm<<<gsz(3l * HH * 128, TPB), TPB, 0, stream>>>(Wx, Vm, WxVmt);

  // ---- sequential scan over time ----
  for (int t = 0; t < TT; ++t) {
    const int* idx = bidx + (size_t)t * EE;

    // prediction + loss (uses PRE-update h)
    k_gather_h<<<gsz((long)EE * HH, TPB), TPB, 0, stream>>>(h, idx, Ah);
    {
      dim3 g(HH / 64, EE / 32);
      k_wmma_gemm<1, 1, 4><<<g, 32, 0, stream>>>(Ah, W1t, b1, nullptr, 0,
                                                 nullptr, Hrelu, HH, HH);
    }
    {
      dim3 g(1, EE / 32);                      // N = 32 -> NT = 2
      k_wmma_gemm<0, 0, 2><<<g, 32, 0, stream>>>(Hrelu, W2t, b2pad, nullptr, 0,
                                                 P, nullptr, 32, HH);
    }
    k_loss<<<gsz((long)EE * (DD - 1), TPB), TPB, 0, stream>>>(P, X, M, obs, t, acc);

    // observation processing
    k_zero_f32<<<1, 128, 0, stream>>>(colsum, 2 * DINN);
    k_observe<<<EE / 4, 256, 0, stream>>>(X, M, idx, obs, t, last_x, last_t, IntB, colsum);
    {
      dim3 g(DINN / 64, EE / 32);
      k_wmma_gemm<4, 0, 4><<<g, 32, 0, stream>>>(IntB, Wgxt, nullptr, nullptr, 0,
                                                 Gx, nullptr, DINN, DINN);
    }
    {
      dim3 g(HH / 64, EE / 32);
      k_wmma_gemm<4, 0, 4><<<g, 32, 0, stream>>>(IntB, Wght, nullptr, nullptr, 0,
                                                 Gh, nullptr, HH, DINN);
    }
    k_xhat<<<gsz((long)EE * DINN, TPB), TPB, 0, stream>>>(X, M, idx, t, last_x, Gx, colsum, AXM);
    k_scale_h<<<gsz((long)EE * HH, TPB), TPB, 0, stream>>>(h, idx, Gh, hsF, hsB);

    // GRU-D cell
    {
      dim3 g(3 * HH / 64, EE / 32);            // pre = [X_hat|M2]@[Wx;Vm] + b
      k_wmma_gemm<0, 0, 4><<<g, 32, 0, stream>>>(AXM, WxVmt, b, nullptr, 0,
                                                 PRE, nullptr, 3 * HH, 128);
    }
    {
      dim3 g(2 * HH / 64, EE / 32);            // rz = sigmoid(pre[:,:2H] + hs@U_rz)
      k_wmma_gemm<2, 0, 4><<<g, 32, 0, stream>>>(hsB, U_rzt, nullptr, PRE, 3 * HH,
                                                 RZ, nullptr, 2 * HH, HH);
    }
    k_rh<<<gsz((long)EE * HH, TPB), TPB, 0, stream>>>(RZ, hsF, rh);
    {
      dim3 g(HH / 64, EE / 32);                // ht = tanh(pre[:,2H:] + (r*hs)@U_h)
      k_wmma_gemm<3, 0, 4><<<g, 32, 0, stream>>>(rh, U_ht, nullptr, PRE + 2 * HH, 3 * HH,
                                                 HT, nullptr, HH, HH);
    }
    k_hupdate<<<gsz((long)EE * HH, TPB), TPB, 0, stream>>>(RZ, hsF, HT, idx, h);
  }

  k_finalize<<<1, 32, 0, stream>>>(acc, out);
}
